// QuantizedAlexNetSmall_58119497450169
// MI455X (gfx1250) — 8-bit matrix operand order repaired and checked by reading, compiled, not run on hardware
//
#include <hip/hip_runtime.h>
#include <hip/hip_bf16.h>
#include <math.h>

typedef unsigned char u8;
typedef unsigned int  u32;
typedef int v8i __attribute__((ext_vector_type(8)));

#define QMAXF 255.0f

static const size_t OFF_B = 50331648u;
static const size_t OFF_W = 61390848u;

__global__ void k_quant_in(const float* __restrict__ x,
                           const float* __restrict__ as_, const float* __restrict__ az_,
                           u8* __restrict__ out, int total)
{
    int idx = blockIdx.x * blockDim.x + threadIdx.x;
    if (idx >= total) return;
    float s0 = as_[0], z0 = az_[0];
    int c  = idx % 3;
    int px = idx / 3;
    int xx = px & 31;
    int yy = (px >> 5) & 31;
    int b  = px >> 10;
    float v = x[((b * 3 + c) * 32 + yy) * 32 + xx];
    float q = rintf(v / s0 + z0);
    q = fminf(fmaxf(q, 0.0f), QMAXF);
    out[idx] = (u8)(int)q;
}

__global__ void k_conv1(const u8* __restrict__ xq, const float* __restrict__ w1,
                        const float* __restrict__ b1,
                        const float* __restrict__ as_, const float* __restrict__ az_,
                        const float* __restrict__ ws_, const float* __restrict__ wz_,
                        u8* __restrict__ out, int total)
{
    int idx = blockIdx.x * blockDim.x + threadIdx.x;
    if (idx >= total) return;
    int oc = idx % 96;
    int px = idx / 96;
    int ox = px & 31;
    int oy = (px >> 5) & 31;
    int b  = px >> 10;
    int zi = (int)az_[0];
    int zw = (int)wz_[0];
    float Msc = as_[0] * ws_[0] / as_[1];
    float zo  = az_[1];
    int acc = 0;
    for (int ci = 0; ci < 3; ++ci)
        for (int kh = 0; kh < 5; ++kh) {
            int iy = oy + kh - 2;
            for (int kw = 0; kw < 5; ++kw) {
                int ix = ox + kw - 2;
                int xv = zi;
                if (iy >= 0 && iy < 32 && ix >= 0 && ix < 32)
                    xv = xq[((b * 32 + iy) * 32 + ix) * 3 + ci];
                int wv = (int)w1[((oc * 3 + ci) * 5 + kh) * 5 + kw];
                acc += (xv - zi) * (wv - zw);
            }
        }
    float fv = (float)rint((double)(acc + (int)b1[oc]) * (double)Msc) + zo;
    fv = fminf(fmaxf(fv, zo), QMAXF);
    out[idx] = (u8)(int)fv;
}

__global__ void k_maxpool(const u8* __restrict__ in, u8* __restrict__ out,
                          int C, int H, int W, int Hout, int Wout, int total)
{
    int idx = blockIdx.x * blockDim.x + threadIdx.x;
    if (idx >= total) return;
    int c  = idx % C;
    int px = idx / C;
    int ox = px % Wout;
    int t  = px / Wout;
    int oy = t % Hout;
    int b  = t / Hout;
    int iy = oy * 2, ix = ox * 2;
    u8 m = 0;
    #pragma unroll
    for (int dy = 0; dy < 3; ++dy)
        #pragma unroll
        for (int dx = 0; dx < 3; ++dx) {
            u8 v = in[((b * H + iy + dy) * W + ix + dx) * C + c];
            m = v > m ? v : m;
        }
    out[idx] = m;
}

__global__ void k_pack_convw(const float* __restrict__ wf, const float* __restrict__ wz_,
                             int layer, u8* __restrict__ w8,
                             int Cin, int KH, int KW, int Keff, int total)
{
    int idx = blockIdx.x * blockDim.x + threadIdx.x;
    if (idx >= total) return;
    int k = idx % Keff;
    int o = idx / Keff;
    int Kreal = Cin * KH * KW;
    int val;
    if (k < Kreal) {
        int grp = k / Cin;
        int ci  = k - grp * Cin;
        int kh  = grp / KW;
        int kw  = grp - kh * KW;
        val = (int)wf[((o * Cin + ci) * KH + kh) * KW + kw];
    } else {
        val = (int)wz_[layer];
    }
    w8[idx] = (u8)val;
}

__global__ void k_pack_fcw(const float* __restrict__ wf, u8* __restrict__ w8, int total)
{
    int idx = blockIdx.x * blockDim.x + threadIdx.x;
    if (idx >= total) return;
    w8[idx] = (u8)(int)wf[idx];
}

__global__ void k_avgpool(const u8* __restrict__ in, u8* __restrict__ out, int total)
{
    int idx = blockIdx.x * blockDim.x + threadIdx.x;
    if (idx >= total) return;
    int c = idx % 256;
    int b = idx / 256;
    u32 s = 0;
    #pragma unroll
    for (int y = 0; y < 3; ++y)
        #pragma unroll
        for (int x = 0; x < 3; ++x)
            s += in[((b * 3 + y) * 3 + x) * 256 + c];
    out[idx] = (u8)(s / 9u);
}

__global__ void __launch_bounds__(256)
k_conv_wmma(const u8* __restrict__ act, const u8* __restrict__ w8,
            const float* __restrict__ bias,
            const float* __restrict__ as_, const float* __restrict__ az_,
            const float* __restrict__ ws_, const float* __restrict__ wz_,
            int layer, u8* __restrict__ out,
            int H, int W, int Cin, int Cout, int KW, int pad,
            int Hout, int Wout, int Ksteps, int Kreal, int Mtotal)
{
    __shared__ u32 ashare[2][256];

    const int tid  = threadIdx.x;
    const int wave = tid >> 5;
    const int lane = tid & 31;
    const int NT = Cout >> 4;
    const int ntBlocks = (NT + 7) >> 3;
    const int mt  = blockIdx.x / ntBlocks;
    const int ntb = blockIdx.x - mt * ntBlocks;
    const int nt  = ntb * 8 + wave;
    const bool ntValid = nt < NT;

    const int   zi   = (int)az_[layer];
    const int   zw   = (int)wz_[layer];
    const float Msc  = as_[layer] * ws_[layer] / as_[layer + 1];
    const float zout = az_[layer + 1];
    const u32   zi4  = 0x01010101u * (u32)(zi & 0xff);

    const int lm = tid >> 4;
    const int lj = tid & 15;
    const int HWo = Hout * Wout;
    const int mrowL = mt * 16 + lm;
    const int bImgL = mrowL / HWo;
    const int remL  = mrowL - bImgL * HWo;
    const int oyL   = remL / Wout;
    const int oxL   = remL - oyL * Wout;

    const int halfK = (lane >> 4);
    const u32* aRowLds0 = &ashare[0][(lane & 15) * 16 + halfK * 2];
    const u32* aRowLds1 = &ashare[1][(lane & 15) * 16 + halfK * 2];
    const int koffB = halfK * 8;
    const int ncol  = nt * 16 + (lane & 15);
    const int Keff  = Ksteps * 64;
    const u8* wrow  = w8 + (size_t)(ntValid ? ncol : 0) * (size_t)Keff;

    v8i acc = {}, accR = {}, accC = {};
    v8i onesT;
    #pragma unroll
    for (int i = 0; i < 8; ++i) ((u32*)&onesT)[i] = 0x01010101u;

    auto stage = [&](int ks, int buf) {
        int k = ks * 64 + lj * 4;
        u32 aw = zi4;
        if (k < Kreal) {
            int grp = k / Cin;
            int ci  = k - grp * Cin;
            int kh  = grp / KW;
            int kw  = grp - kh * KW;
            int iy  = oyL + kh - pad;
            int ix  = oxL + kw - pad;
            if (iy >= 0 && iy < H && ix >= 0 && ix < W) {
                size_t ofs = ((size_t)((bImgL * H + iy) * W + ix)) * (size_t)Cin + (size_t)ci;
                aw = *(const u32*)(act + ofs);
            }
        }
        ashare[buf][lm * 16 + lj] = aw;
    };

    stage(0, 0);
    __syncthreads();

    for (int ks = 0; ks < Ksteps; ++ks) {
        const int cur = ks & 1;
        if (ks + 1 < Ksteps) stage(ks + 1, cur ^ 1);
        if (ntValid) {
            const u32* arow = cur ? aRowLds1 : aRowLds0;
            const int kb = ks * 64;
            v8i A, Bm;
            #pragma unroll
            for (int p = 0; p < 4; ++p) {
                ((u32*)&A)[2 * p]     = arow[p * 4];
                ((u32*)&A)[2 * p + 1] = arow[p * 4 + 1];
            }
            #pragma unroll
            for (int v = 0; v < 8; ++v) {
                int kbv = kb + ((v >> 1) * 16) + ((v & 1) * 4) + koffB;
                ((u32*)&Bm)[v] = *(const u32*)(wrow + kbv);
            }
            __builtin_prefetch(wrow + kb + 64, 0, 1);
            acc  = __builtin_amdgcn_wmma_i32_16x16x64_iu8(false, A, false, Bm,    acc,  false, false);
            accR = __builtin_amdgcn_wmma_i32_16x16x64_iu8(false, A, false, onesT, accR, false, false);
            accC = __builtin_amdgcn_wmma_i32_16x16x64_iu8(false, onesT, false, Bm, accC, false, false);
        }
        __syncthreads();
    }

    if (!ntValid) return;

    const float bz     = bias[ncol];
    const int   kconst = Keff * zi * zw;
    #pragma unroll
    for (int v = 0; v < 8; ++v) {
        int mm = mt * 16 + halfK * 8 + v;
        int a  = acc[v] - zw * accR[v] - zi * accC[v] + kconst + (int)bz;
        float fv = (float)rint((double)a * (double)Msc) + zout;
        fv = fminf(fmaxf(fv, zout), QMAXF);
        int b2  = mm / HWo;
        int r2  = mm - b2 * HWo;
        int oy2 = r2 / Wout;
        int ox2 = r2 - oy2 * Wout;
        out[((size_t)((b2 * Hout + oy2) * Wout + ox2)) * (size_t)Cout + (size_t)ncol] = (u8)(int)fv;
    }
}

__global__ void __launch_bounds__(256)
k_fc_wmma(const u8* __restrict__ act, const u8* __restrict__ w8,
          const float* __restrict__ bias,
          const float* __restrict__ as_, const float* __restrict__ az_,
          const float* __restrict__ ws_, const float* __restrict__ wz_,
          int layer, int M, int K, int Npad, int Nreal, int relu, int final_f32,
          u8* __restrict__ out_u8, float* __restrict__ out_f32)
{
    __shared__ u32 ashare[2][256];

    const int tid  = threadIdx.x;
    const int wave = tid >> 5;
    const int lane = tid & 31;
    const int NT = Npad >> 4;
    const int ntBlocks = (NT + 7) >> 3;
    const int mt  = blockIdx.x / ntBlocks;
    const int ntb = blockIdx.x - mt * ntBlocks;
    const int nt  = ntb * 8 + wave;
    const bool ntValid = nt < NT;

    const int   zi   = (int)az_[layer];
    const int   zw   = (int)wz_[layer];
    const float Msc  = as_[layer] * ws_[layer] / as_[layer + 1];
    const float zout = az_[layer + 1];

    const int lm = tid >> 4;
    const int lj = tid & 15;
    const u8* arowG = act + (size_t)(mt * 16 + lm) * (size_t)K;

    const int halfK = (lane >> 4);
    const u32* aRowLds0 = &ashare[0][(lane & 15) * 16 + halfK * 2];
    const u32* aRowLds1 = &ashare[1][(lane & 15) * 16 + halfK * 2];
    const int koffB = halfK * 8;
    const int ncol  = nt * 16 + (lane & 15);
    const int ncl   = ncol < Nreal ? ncol : (Nreal - 1);
    const u8* wrow  = w8 + (size_t)(ntValid ? ncl : 0) * (size_t)K;

    v8i acc = {}, accR = {}, accC = {};
    v8i onesT;
    #pragma unroll
    for (int i = 0; i < 8; ++i) ((u32*)&onesT)[i] = 0x01010101u;

    const int Ksteps = K >> 6;
    ashare[0][lm * 16 + lj] = *(const u32*)(arowG + lj * 4);
    __syncthreads();

    for (int ks = 0; ks < Ksteps; ++ks) {
        const int cur = ks & 1;
        if (ks + 1 < Ksteps)
            ashare[cur ^ 1][lm * 16 + lj] = *(const u32*)(arowG + (ks + 1) * 64 + lj * 4);
        if (ntValid) {
            const u32* arow = cur ? aRowLds1 : aRowLds0;
            const int kb = ks * 64;
            v8i A, Bm;
            #pragma unroll
            for (int p = 0; p < 4; ++p) {
                ((u32*)&A)[2 * p]     = arow[p * 4];
                ((u32*)&A)[2 * p + 1] = arow[p * 4 + 1];
            }
            #pragma unroll
            for (int v = 0; v < 8; ++v) {
                int kbv = kb + ((v >> 1) * 16) + ((v & 1) * 4) + koffB;
                ((u32*)&Bm)[v] = *(const u32*)(wrow + kbv);
            }
            __builtin_prefetch(wrow + kb + 64, 0, 1);
            acc  = __builtin_amdgcn_wmma_i32_16x16x64_iu8(false, A, false, Bm,    acc,  false, false);
            accR = __builtin_amdgcn_wmma_i32_16x16x64_iu8(false, A, false, onesT, accR, false, false);
            accC = __builtin_amdgcn_wmma_i32_16x16x64_iu8(false, onesT, false, Bm, accC, false, false);
        }
        __syncthreads();
    }

    if (!ntValid) return;

    const float bz     = bias[ncl];
    const int   kconst = K * zi * zw;
    const float lo     = relu ? zout : 0.0f;
    #pragma unroll
    for (int v = 0; v < 8; ++v) {
        int mm = mt * 16 + halfK * 8 + v;
        int a  = acc[v] - zw * accR[v] - zi * accC[v] + kconst + (int)bz;
        float fv = (float)rint((double)a * (double)Msc) + zout;
        fv = fminf(fmaxf(fv, lo), QMAXF);
        if (ncol < Nreal) {
            if (final_f32)
                out_f32[(size_t)mm * (size_t)Nreal + (size_t)ncol] = fv;
            else
                out_u8[(size_t)mm * (size_t)Nreal + (size_t)ncol] = (u8)(int)fv;
        }
    }
}

extern "C" void kernel_launch(void* const* d_in, const int* in_sizes, int n_in,
                              void* d_out, int out_size, void* d_ws, size_t ws_size,
                              hipStream_t stream)
{
    (void)in_sizes; (void)n_in; (void)out_size; (void)ws_size;
    const float* x  = (const float*)d_in[0];
    const float* Wt[8];
    const float* Bs[8];
    for (int i = 0; i < 8; ++i) Wt[i] = (const float*)d_in[1 + i];
    for (int i = 0; i < 8; ++i) Bs[i] = (const float*)d_in[9 + i];
    const float* AS = (const float*)d_in[17];
    const float* AZ = (const float*)d_in[18];
    const float* WS = (const float*)d_in[19];
    const float* WZ = (const float*)d_in[20];
    float* outp = (float*)d_out;

    u8* base = (u8*)d_ws;
    u8* bufA = base;
    u8* bufB = base + OFF_B;
    u8* bufW = base + OFF_W;

    int t;

    t = 512 * 32 * 32 * 3;
    k_quant_in<<<(t + 255) / 256, 256, 0, stream>>>(x, AS, AZ, bufB, t);

    t = 512 * 32 * 32 * 96;
    k_conv1<<<(t + 255) / 256, 256, 0, stream>>>(bufB, Wt[0], Bs[0], AS, AZ, WS, WZ, bufA, t);

    t = 512 * 15 * 15 * 96;
    k_maxpool<<<(t + 255) / 256, 256, 0, stream>>>(bufA, bufB, 96, 32, 32, 15, 15, t);

    t = 256 * 2432;
    k_pack_convw<<<(t + 255) / 256, 256, 0, stream>>>(Wt[1], WZ, 1, bufW, 96, 5, 5, 2432, t);
    {
        int Mtot = 512 * 15 * 15;
        int blocks = (Mtot / 16) * ((256 / 16 + 7) / 8);
        k_conv_wmma<<<blocks, 256, 0, stream>>>(bufB, bufW, Bs[1], AS, AZ, WS, WZ, 1,
            bufA, 15, 15, 96, 256, 5, 2, 15, 15, 38, 2400, Mtot);
    }

    t = 512 * 7 * 7 * 256;
    k_maxpool<<<(t + 255) / 256, 256, 0, stream>>>(bufA, bufB, 256, 15, 15, 7, 7, t);

    t = 384 * 2304;
    k_pack_convw<<<(t + 255) / 256, 256, 0, stream>>>(Wt[2], WZ, 2, bufW, 256, 3, 3, 2304, t);
    {
        int Mtot = 512 * 7 * 7;
        int blocks = (Mtot / 16) * ((384 / 16 + 7) / 8);
        k_conv_wmma<<<blocks, 256, 0, stream>>>(bufB, bufW, Bs[2], AS, AZ, WS, WZ, 2,
            bufA, 7, 7, 256, 384, 3, 1, 7, 7, 36, 2304, Mtot);
    }

    t = 384 * 3456;
    k_pack_convw<<<(t + 255) / 256, 256, 0, stream>>>(Wt[3], WZ, 3, bufW, 384, 3, 3, 3456, t);
    {
        int Mtot = 512 * 7 * 7;
        int blocks = (Mtot / 16) * ((384 / 16 + 7) / 8);
        k_conv_wmma<<<blocks, 256, 0, stream>>>(bufA, bufW, Bs[3], AS, AZ, WS, WZ, 3,
            bufB, 7, 7, 384, 384, 3, 1, 7, 7, 54, 3456, Mtot);
    }

    t = 256 * 3456;
    k_pack_convw<<<(t + 255) / 256, 256, 0, stream>>>(Wt[4], WZ, 4, bufW, 384, 3, 3, 3456, t);
    {
        int Mtot = 512 * 7 * 7;
        int blocks = (Mtot / 16) * ((256 / 16 + 7) / 8);
        k_conv_wmma<<<blocks, 256, 0, stream>>>(bufB, bufW, Bs[4], AS, AZ, WS, WZ, 4,
            bufA, 7, 7, 384, 256, 3, 1, 7, 7, 54, 3456, Mtot);
    }

    t = 512 * 3 * 3 * 256;
    k_maxpool<<<(t + 255) / 256, 256, 0, stream>>>(bufA, bufB, 256, 7, 7, 3, 3, t);

    t = 512 * 256;
    k_avgpool<<<(t + 255) / 256, 256, 0, stream>>>(bufB, bufA, t);

    t = 4096 * 256;
    k_pack_fcw<<<(t + 255) / 256, 256, 0, stream>>>(Wt[5], bufW, t);
    {
        int blocks = (512 / 16) * ((4096 / 16 + 7) / 8);
        k_fc_wmma<<<blocks, 256, 0, stream>>>(bufA, bufW, Bs[5], AS, AZ, WS, WZ,
            5, 512, 256, 4096, 4096, 1, 0, bufB, nullptr);
    }

    t = 4096 * 4096;
    k_pack_fcw<<<(t + 255) / 256, 256, 0, stream>>>(Wt[6], bufW, t);
    {
        int blocks = (512 / 16) * ((4096 / 16 + 7) / 8);
        k_fc_wmma<<<blocks, 256, 0, stream>>>(bufB, bufW, Bs[6], AS, AZ, WS, WZ,
            6, 512, 4096, 4096, 4096, 1, 0, bufA, nullptr);
    }

    t = 10 * 4096;
    k_pack_fcw<<<(t + 255) / 256, 256, 0, stream>>>(Wt[7], bufW, t);
    {
        int blocks = (512 / 16) * 1;
        k_fc_wmma<<<blocks, 256, 0, stream>>>(bufA, bufW, Bs[7], AS, AZ, WS, WZ,
            7, 512, 4096, 16, 10, 0, 1, nullptr, outp);
    }
}
